// AttentionBlock_90494960926893
// MI455X (gfx1250) — matrix operand order repaired and checked by reading, compiled, not run on hardware
//
#include <hip/hip_runtime.h>
#include <hip/hip_bf16.h>
#include <math.h>

// Problem constants (from the reference): N=16, C=512, S=4096.
#define BN 16
#define BC 512
#define BS 4096

// ---- CDNA5 vector types for WMMA -------------------------------------------
typedef __attribute__((ext_vector_type(16))) __bf16        v16bf;
typedef __attribute__((ext_vector_type(8)))  float         v8f;
typedef __attribute__((ext_vector_type(4)))  unsigned int  u32x4;

union FragU {
    v16bf v;      // 16 bf16 = 32 bytes
    u32x4 q[2];   // two 16-byte chunks for aligned LDS reads
};

// Two fp32 -> packed 2x bf16 (round-to-nearest, ties away from zero).
// 3 VALU ops total: 2x v_add_nc_u32 + 1x v_perm_b32.
//   result[15:0]  = (bits(a)+0x8000)[31:16]   (selector bytes 2,3 of src1)
//   result[31:16] = (bits(b)+0x8000)[31:16]   (selector bytes 6,7 of src0)
__device__ __forceinline__ unsigned int pk2(float a, float b) {
    union { float f; unsigned int u; } xa, xb;
    xa.f = a; xb.f = b;
    const unsigned int ra = xa.u + 0x8000u;
    const unsigned int rb = xb.u + 0x8000u;
    return __builtin_amdgcn_perm(rb, ra, 0x07060302u);
}

// Pack 8 floats into 8 bf16 (one 16B chunk)
__device__ __forceinline__ u32x4 pack8(const float* f) {
    u32x4 r;
    r.x = pk2(f[0], f[1]);
    r.y = pk2(f[2], f[3]);
    r.z = pk2(f[4], f[5]);
    r.w = pk2(f[6], f[7]);
    return r;
}

// ---------------------------------------------------------------------------
// Kernel 1:  sim[n, c, d] = scale * sum_s key[n,c,s] * query[n,d,s]
// GEMM: M = c (64/block), N = d (128/block), K = s (chunks of 32).
// A = key (row-major along k), B^T = query (row-major along k) -> both tiles
// stage identically into LDS as bf16.
// 256 threads = 8 waves laid out 2 (M) x 4 (N); each wave: 32x32 output.
// ---------------------------------------------------------------------------
__global__ void __launch_bounds__(256)
sim_gemm_kernel(const float* __restrict__ key,
                const float* __restrict__ query,
                float* __restrict__ sim, float scale) {
    __shared__ __align__(16) unsigned short As[64 * 32];   // [m][k] bf16
    __shared__ __align__(16) unsigned short Bs[128 * 32];  // [n][k] bf16

    const int tid   = threadIdx.x;
    const int lane  = tid & 31;
    const int wave  = tid >> 5;
    const int wm    = (wave >> 2) * 32;   // wave M offset within block tile
    const int wn    = (wave & 3)  * 32;   // wave N offset within block tile
    const int l15   = lane & 15;
    const int hi    = lane >> 4;

    const int n0    = blockIdx.x * 128;   // d tile
    const int m0    = blockIdx.y * 64;    // c tile
    const int batch = blockIdx.z;

    const float* kbase = key   + ((size_t)batch * BC) * BS;
    const float* qbase = query + ((size_t)batch * BC) * BS;

    // staging assignments
    const int a_row = tid >> 2, a_cb = (tid & 3) * 8;    // 64 rows x 32 k
    const int b_row = tid >> 1, b_cb = (tid & 1) * 16;   // 128 rows x 32 k

    v8f acc[2][2];
    acc[0][0] = {}; acc[0][1] = {}; acc[1][0] = {}; acc[1][1] = {};

    const int NK = BS / 32;
    for (int kb = 0; kb < NK; ++kb) {
        // ---- stage A (key) tile: fp32 -> bf16 ----
        {
            const float* src = kbase + (size_t)(m0 + a_row) * BS + kb * 32 + a_cb;
            float f[8];
            #pragma unroll
            for (int i = 0; i < 8; ++i) f[i] = src[i];
            *(u32x4*)(As + a_row * 32 + a_cb) = pack8(f);
            if (kb + 1 < NK) __builtin_prefetch(src + 32, 0, 3);
        }
        // ---- stage B (query) tile ----
        {
            const float* src = qbase + (size_t)(n0 + b_row) * BS + kb * 32 + b_cb;
            float f[16];
            #pragma unroll
            for (int i = 0; i < 16; ++i) f[i] = src[i];
            *(u32x4*)(Bs + b_row * 32 + b_cb)     = pack8(f);
            *(u32x4*)(Bs + b_row * 32 + b_cb + 8) = pack8(f + 8);
            if (kb + 1 < NK) __builtin_prefetch(src + 32, 0, 3);
        }
        __syncthreads();

        // ---- fragments + 4 WMMAs ----
        FragU a[2], b[2];
        #pragma unroll
        for (int mi = 0; mi < 2; ++mi) {
            const u32x4* row = (const u32x4*)(As + (wm + mi * 16 + l15) * 32);
            a[mi].q[0] = row[2 * hi];    // K = hi*16 .. +7
            a[mi].q[1] = row[2 * hi + 1];    // K = hi*16 + 8 .. +15
        }
        #pragma unroll
        for (int ni = 0; ni < 2; ++ni) {
            const u32x4* row = (const u32x4*)(Bs + (wn + ni * 16 + l15) * 32);
            b[ni].q[0] = row[2 * hi];    // K = hi*16 .. +15 (contiguous 32B)
            b[ni].q[1] = row[2 * hi + 1];
        }
        #pragma unroll
        for (int mi = 0; mi < 2; ++mi)
            #pragma unroll
            for (int ni = 0; ni < 2; ++ni)
                acc[mi][ni] = __builtin_amdgcn_wmma_f32_16x16x32_bf16(
                    false, a[mi].v, false, b[ni].v,
                    (short)0, acc[mi][ni], false, false);
        __syncthreads();
    }

    // ---- store C (scaled), per documented 16x16 f32 layout ----
    #pragma unroll
    for (int mi = 0; mi < 2; ++mi) {
        #pragma unroll
        for (int ni = 0; ni < 2; ++ni) {
            const int col = n0 + wn + ni * 16 + l15;
            #pragma unroll
            for (int v = 0; v < 8; ++v) {
                const int row = m0 + wm + mi * 16 + hi * 8 + v;
                sim[((size_t)batch * BC + row) * BC + col] = acc[mi][ni][v] * scale;
            }
        }
    }
}

// ---------------------------------------------------------------------------
// Kernel 2: softmax over axis c (dim=1) of sim[n, c, d], in place.
// One thread per (n, d) column; adjacent threads -> adjacent addresses.
// ---------------------------------------------------------------------------
__global__ void __launch_bounds__(256)
softmax_col_kernel(float* __restrict__ sim) {
    const int idx = blockIdx.x * 256 + threadIdx.x;  // 0 .. N*C-1
    const int n = idx >> 9;        // / C
    const int d = idx & (BC - 1);  // % C
    float* col = sim + ((size_t)n * BC) * BC + d;    // stride BC over c

    float m = -INFINITY, s = 0.0f;
    for (int c = 0; c < BC; ++c) {
        const float x = col[(size_t)c * BC];
        const float mn = fmaxf(m, x);
        s = s * __expf(m - mn) + __expf(x - mn);
        m = mn;
    }
    const float inv = 1.0f / s;
    for (int c = 0; c < BC; ++c) {
        const float x = col[(size_t)c * BC];
        col[(size_t)c * BC] = __expf(x - m) * inv;
    }
}

// ---------------------------------------------------------------------------
// Kernel 3:  out[n, c, s] = sum_d P[n,c,d] * value[n,d,s]
// GEMM: M = c (64/block), N = s (128/block), K = d (chunks of 32).
// A = P (row-major along k).  B[k=d, n=s] = value[d][s]: stage V tile
// transposed in LDS so per-lane B fragments are contiguous in K.
// Transpose staging packs (d, d+1) pairs -> one 32-bit DS store per 2 elems.
// ---------------------------------------------------------------------------
__global__ void __launch_bounds__(256)
ctx_gemm_kernel(const float* __restrict__ P,
                const float* __restrict__ value,
                float* __restrict__ out) {
    __shared__ __align__(16) unsigned short As[64 * 32];   // [m][k] bf16
    __shared__ __align__(16) unsigned short Bt[128 * 32];  // [n=s][k=d] bf16

    const int tid   = threadIdx.x;
    const int lane  = tid & 31;
    const int wave  = tid >> 5;
    const int wm    = (wave >> 2) * 32;
    const int wn    = (wave & 3)  * 32;
    const int l15   = lane & 15;
    const int hi    = lane >> 4;

    const int n0    = blockIdx.x * 128;   // s tile
    const int m0    = blockIdx.y * 64;    // c tile
    const int batch = blockIdx.z;

    const float* pbase = P     + ((size_t)batch * BC) * BC;
    const float* vbase = value + ((size_t)batch * BC) * BS;

    const int a_row = tid >> 2, a_cb = (tid & 3) * 8;        // 64 x 32 (k=d)
    const int v_d0  = (tid & 15) * 2, v_sb = (tid >> 4) * 8; // 2 d-rows x 8 s

    v8f acc[2][2];
    acc[0][0] = {}; acc[0][1] = {}; acc[1][0] = {}; acc[1][1] = {};

    const int NK = BC / 32;
    for (int kb = 0; kb < NK; ++kb) {
        // ---- stage A (P) tile ----
        {
            const float* src = pbase + (size_t)(m0 + a_row) * BC + kb * 32 + a_cb;
            float f[8];
            #pragma unroll
            for (int i = 0; i < 8; ++i) f[i] = src[i];
            *(u32x4*)(As + a_row * 32 + a_cb) = pack8(f);
            if (kb + 1 < NK) __builtin_prefetch(src + 32, 0, 3);
        }
        // ---- stage B (value) tile, transposed: Bt[s_local][d_local] ----
        {
            const float* src0 = vbase + (size_t)(kb * 32 + v_d0)     * BS + n0 + v_sb;
            const float* src1 = vbase + (size_t)(kb * 32 + v_d0 + 1) * BS + n0 + v_sb;
            float f0[8], f1[8];
            #pragma unroll
            for (int i = 0; i < 8; ++i) f0[i] = src0[i];
            #pragma unroll
            for (int i = 0; i < 8; ++i) f1[i] = src1[i];
            #pragma unroll
            for (int i = 0; i < 8; ++i)
                *(unsigned int*)(Bt + (v_sb + i) * 32 + v_d0) = pk2(f0[i], f1[i]);
            if (kb + 1 < NK)
                __builtin_prefetch(vbase + (size_t)(kb * 32 + 32 + v_d0) * BS + n0 + v_sb, 0, 3);
        }
        __syncthreads();

        FragU a[2], b[2];
        #pragma unroll
        for (int mi = 0; mi < 2; ++mi) {
            const u32x4* row = (const u32x4*)(As + (wm + mi * 16 + l15) * 32);
            a[mi].q[0] = row[2 * hi];
            a[mi].q[1] = row[2 * hi + 1];
        }
        #pragma unroll
        for (int ni = 0; ni < 2; ++ni) {
            const u32x4* row = (const u32x4*)(Bt + (wn + ni * 16 + l15) * 32);
            b[ni].q[0] = row[2 * hi];
            b[ni].q[1] = row[2 * hi + 1];
        }
        #pragma unroll
        for (int mi = 0; mi < 2; ++mi)
            #pragma unroll
            for (int ni = 0; ni < 2; ++ni)
                acc[mi][ni] = __builtin_amdgcn_wmma_f32_16x16x32_bf16(
                    false, a[mi].v, false, b[ni].v,
                    (short)0, acc[mi][ni], false, false);
        __syncthreads();
    }

    #pragma unroll
    for (int mi = 0; mi < 2; ++mi) {
        #pragma unroll
        for (int ni = 0; ni < 2; ++ni) {
            const int col = n0 + wn + ni * 16 + l15;
            #pragma unroll
            for (int v = 0; v < 8; ++v) {
                const int row = m0 + wm + mi * 16 + hi * 8 + v;
                out[((size_t)batch * BC + row) * BS + col] = acc[mi][ni][v];
            }
        }
    }
}

// ---------------------------------------------------------------------------
// Inputs (setup_inputs order): key [N,C,S] f32, value [N,C,S] f32,
// query [N,C,S] f32. Output: context [N,C,S] f32.
// Workspace: sim/P buffer = N*C*C*4 = 16 MB.
// ---------------------------------------------------------------------------
extern "C" void kernel_launch(void* const* d_in, const int* in_sizes, int n_in,
                              void* d_out, int out_size, void* d_ws, size_t ws_size,
                              hipStream_t stream) {
    const float* key   = (const float*)d_in[0];
    const float* value = (const float*)d_in[1];
    const float* query = (const float*)d_in[2];
    float* out = (float*)d_out;
    float* sim = (float*)d_ws;  // 16 MB: [N, C, C]

    const float scale = 0.044194173824159216f;  // 512^-0.5

    dim3 g1(BC / 128, BC / 64, BN);   // sim GEMM: d tiles x c tiles x batch
    sim_gemm_kernel<<<g1, 256, 0, stream>>>(key, query, sim, scale);

    dim3 g2((BN * BC) / 256);         // one thread per (n, d) column
    softmax_col_kernel<<<g2, 256, 0, stream>>>(sim);

    dim3 g3(BS / 128, BC / 64, BN);   // ctx GEMM: s tiles x c tiles x batch
    ctx_gemm_kernel<<<g3, 256, 0, stream>>>(sim, value, out);
}
